// CustomMultiheadAttention_67473936220854
// MI455X (gfx1250) — matrix operand order repaired and checked by reading, not compiled, not run on hardware
//
#include <hip/hip_runtime.h>
#include <hip/hip_bf16.h>

// ---------------------------------------------------------------------------
// CDNA5 (gfx1250) fused multi-head attention with relative-position bias.
//   scores = Q (K*scale + rel_sum)^T   -> rel-pos folds into K' projection
// All GEMMs run on v_wmma_f32_16x16x32_bf16 (wave32); K'/V staging uses
// GLOBAL_LOAD_ASYNC_TO_LDS (ASYNCcnt) when the toolchain exposes it.
// ---------------------------------------------------------------------------

typedef __bf16 bf16;
typedef __attribute__((ext_vector_type(16))) __bf16 v16bf;
typedef __attribute__((ext_vector_type(8)))  float  v8f;
typedef __attribute__((ext_vector_type(2)))  int    v2i;

#define WMMA_BF16(A, Bm, C) \
  __builtin_amdgcn_wmma_f32_16x16x32_bf16(false, (A), false, (Bm), (short)0, (C), false, false)

static constexpr int S_ = 1024, Bb = 8, E_ = 1024, H_ = 16, HD = 64;
static constexpr int ROWS = S_ * Bb;     // 8192 rows for (S*B, E) GEMMs
static constexpr float SCALE = 0.125f;   // 1/sqrt(64)

// ---- CDNA5 async global->LDS copy (ASYNCcnt), with portable fallback ------
#if __has_builtin(__builtin_amdgcn_global_load_async_to_lds_b64) && \
    __has_builtin(__builtin_amdgcn_s_wait_asynccnt)
#define USE_ASYNC_LDS 1
#else
#define USE_ASYNC_LDS 0
#endif

__device__ inline void cp8_g2l(const bf16* g, bf16* l) {
#if USE_ASYNC_LDS
  // Builtin signature (from clang diagnostic): (v2i AS1*, v2i AS3*, i32, i32).
  // LDS generic pointers keep the byte offset in addr[31:0] (ISA 10.2).
  __builtin_amdgcn_global_load_async_to_lds_b64(
      (__attribute__((address_space(1))) v2i*)(unsigned long long)(g),
      (__attribute__((address_space(3))) v2i*)(unsigned int)(unsigned long long)(l),
      0, 0);
#else
  *(uint2*)l = *(const uint2*)g;
#endif
}
__device__ inline void async_fence_acquire() {
#if USE_ASYNC_LDS
  __builtin_amdgcn_s_wait_asynccnt(0);
#endif
  __syncthreads();
}

// ---- fragment loaders (layouts per CDNA5 ISA 7.12.2) -----------
// A 16x32 bf16: lane L holds row M=L%16; element s -> K = hf*16 + s,
// hf = L/16 (contiguous order, matching load_b_bf16).
__device__ inline v16bf load_a_bf16(const bf16* row, int kbase, int hf) {
  union { v16bf v; bf16 h[16]; } u;
  const bf16* p0 = row + kbase + hf * 16;
  const bf16* p1 = row + kbase + hf * 16 + 8;
#pragma unroll
  for (int i = 0; i < 8; ++i) { u.h[i] = p0[i]; u.h[8 + i] = p1[i]; }
  return u.v;
}
// B 32x16 bf16: lane L holds col N=L%16; lanes 0-15 K=0..15, lanes 16-31
// K=16..31.  Caller passes pointer to its 16 consecutive K values.
__device__ inline v16bf load_b_bf16(const bf16* p) {
  union { v16bf v; bf16 h[16]; } u;
#pragma unroll
  for (int i = 0; i < 16; ++i) u.h[i] = p[i];
  return u.v;
}
__device__ inline uint2 pack4_bf16(float4 f) {
  union { uint2 u; bf16 h[4]; } pk;
  pk.h[0] = (bf16)f.x; pk.h[1] = (bf16)f.y; pk.h[2] = (bf16)f.z; pk.h[3] = (bf16)f.w;
  return pk.u;
}

// ---------------------------------------------------------------------------
// Kernel 0: rel_sum[l,d] = sum_i table[clip(i-l+127,0,254), d]
// ---------------------------------------------------------------------------
__global__ void rel_sum_kernel(const float* __restrict__ T, float* __restrict__ out) {
  const int l = blockIdx.x;      // 0..1023
  const int d = threadIdx.x;     // 0..63
  const float c0   = (float)max(0, l - 126);
  const float c254 = (float)max(0, 897 - l);
  float acc = c0 * T[d] + c254 * T[254 * HD + d];
  for (int t = 1; t <= 253; ++t) {
    const int i = t + l - 127;
    if (i >= 0 && i < S_) acc += T[t * HD + d];
  }
  out[l * HD + d] = acc;
}

// ---------------------------------------------------------------------------
// Kernel 1: LDS-tiled GEMM  C = X @ W^T + bias.
// Block tile 128(m) x 64(n), 8 waves as 4x2, each wave a 32x32 sub-tile
// (4 WMMA accumulators).  Per k-step(32) the block stages a 128x32 A-panel
// and 64x32 B-panel as bf16 in LDS (padded stride 36 -> conflict-free
// 16-lane fragment reads).
// MODE 0: Q  -> bf16 [b,h,s,d]
// MODE 1: K' -> bf16 [b,h,s,d],  K' = SCALE*K + rel_sum[s,d]
// MODE 2: V  -> bf16 [b,h,d,s]   (transposed for contiguous PV B-fragments)
// MODE 3: out-projection: A is bf16 (B*S,E), output fp32 in (S,B,E)
// ---------------------------------------------------------------------------
template <int MODE>
__global__ __launch_bounds__(256, 2)
void gemm_kernel(const float* __restrict__ Xf, const bf16* __restrict__ Xb,
                 const float* __restrict__ W, const float* __restrict__ bias,
                 const float* __restrict__ rel_sum,
                 bf16* __restrict__ Outb, float* __restrict__ Outf) {
  __shared__ __align__(16) bf16 sX[128][36];
  __shared__ __align__(16) bf16 sW[64][36];

  const int tid = threadIdx.x;
  const int lane = tid & 31, ln = lane & 15, hf = lane >> 4, wv = tid >> 5;
  const int wrow = wv >> 1, wcol = wv & 1;
  const int m0 = blockIdx.x * 128;
  const int n0 = blockIdx.y * 64;

  v8f acc[2][2];
#pragma unroll
  for (int mt = 0; mt < 2; ++mt)
#pragma unroll
    for (int nt = 0; nt < 2; ++nt) { v8f z = {}; acc[mt][nt] = z; }

  for (int kb = 0; kb < E_; kb += 32) {
    __syncthreads();
    // ---- stage A panel: 128 rows x 32 k (1024 4-element chunks) ----
#pragma unroll
    for (int i = 0; i < 4; ++i) {
      const int idx = i * 256 + tid;
      const int row = idx >> 3, c = (idx & 7) * 4;
      if (MODE == 3) {
        *(uint2*)&sX[row][c] = *(const uint2*)(Xb + (size_t)(m0 + row) * E_ + kb + c);
      } else {
        const float4 f = *(const float4*)(Xf + (size_t)(m0 + row) * E_ + kb + c);
        *(uint2*)&sX[row][c] = pack4_bf16(f);
      }
    }
    // ---- stage B panel: 64 rows x 32 k (512 chunks) ----
#pragma unroll
    for (int i = 0; i < 2; ++i) {
      const int idx = i * 256 + tid;
      const int row = idx >> 3, c = (idx & 7) * 4;
      const float4 f = *(const float4*)(W + (size_t)(n0 + row) * E_ + kb + c);
      *(uint2*)&sW[row][c] = pack4_bf16(f);
    }
    __syncthreads();

    const v16bf a0 = load_a_bf16(&sX[wrow * 32 + ln][0], 0, hf);
    const v16bf a1 = load_a_bf16(&sX[wrow * 32 + 16 + ln][0], 0, hf);
    const v16bf b0 = load_b_bf16(&sW[wcol * 32 + ln][hf * 16]);
    const v16bf b1 = load_b_bf16(&sW[wcol * 32 + 16 + ln][hf * 16]);
    acc[0][0] = WMMA_BF16(a0, b0, acc[0][0]);
    acc[0][1] = WMMA_BF16(a0, b1, acc[0][1]);
    acc[1][0] = WMMA_BF16(a1, b0, acc[1][0]);
    acc[1][1] = WMMA_BF16(a1, b1, acc[1][1]);
  }

  // ---- epilogue ----
#pragma unroll
  for (int mt = 0; mt < 2; ++mt)
#pragma unroll
    for (int nt = 0; nt < 2; ++nt) {
      const int n = n0 + wcol * 32 + nt * 16 + ln;
      const float bn = bias[n];
      const int h = n >> 6, d = n & 63;
#pragma unroll
      for (int v = 0; v < 8; ++v) {
        float val = acc[mt][nt][v] + bn;
        const int r = m0 + wrow * 32 + mt * 16 + v + 8 * hf;
        if (MODE == 3) {                       // r = batch*S + s
          const int batch = r >> 10, s = r & 1023;
          Outf[(size_t)(s * Bb + batch) * E_ + n] = val;
        } else {                               // r = s*B + batch
          const int batch = r & 7, s = r >> 3;
          if (MODE == 1) val = SCALE * val + rel_sum[s * HD + d];
          if (MODE == 2)
            Outb[((size_t)(batch * H_ + h) * HD + d) * S_ + s] = (bf16)val;
          else
            Outb[((size_t)(batch * H_ + h) * S_ + s) * HD + d] = (bf16)val;
        }
      }
    }
}

// ---------------------------------------------------------------------------
// Kernel 2: flash attention.  Block = 8 waves on one (b,h); each wave owns a
// 16-row Q tile.  32-key K'/V chunks staged in LDS via async-to-LDS copies;
// online softmax with 16-lane butterflies; P bounced through per-wave LDS
// (C-layout -> A-layout).  LDS strides padded for conflict-free reads.
// ---------------------------------------------------------------------------
__global__ __launch_bounds__(256, 2)
void attn_kernel(const bf16* __restrict__ Q, const bf16* __restrict__ K,
                 const bf16* __restrict__ Vt, bf16* __restrict__ AO) {
  __shared__ __align__(16) bf16 sK[32][76];      // [key][d]   stride 38 dw
  __shared__ __align__(16) bf16 sV[64][44];      // [d][key]   stride 22 dw
  __shared__ __align__(16) bf16 sP[8][16][36];   // per-wave P stride 18 dw

  const int tid = threadIdx.x;
  const int lane = tid & 31, ln = lane & 15, hf = lane >> 4, wv = tid >> 5;
  const int bh = blockIdx.x >> 3;                // b*16 + h
  const int q0 = (blockIdx.x & 7) * 128 + wv * 16;

  const bf16* Qb = Q  + (size_t)bh * S_ * HD;
  const bf16* Kb = K  + (size_t)bh * S_ * HD;
  const bf16* Vb = Vt + (size_t)bh * HD * S_;

  const bf16* Qrow = Qb + (size_t)(q0 + ln) * HD;
  const v16bf aq0 = load_a_bf16(Qrow, 0, hf);
  const v16bf aq1 = load_a_bf16(Qrow, 32, hf);

  float mrow[8], lrow[8];
  v8f accO[4];
#pragma unroll
  for (int v = 0; v < 8; ++v) { mrow[v] = -__builtin_inff(); lrow[v] = 0.f; }
#pragma unroll
  for (int t = 0; t < 4; ++t) { v8f z = {}; accO[t] = z; }

  for (int key0 = 0; key0 < S_; key0 += 32) {
    __syncthreads();
    // sK: 32 keys x 64 d  = 512 8-byte chunks
#pragma unroll
    for (int i = 0; i < 2; ++i) {
      const int idx = i * 256 + tid;
      const int kk = idx >> 4, c = (idx & 15) * 4;
      cp8_g2l(Kb + (size_t)(key0 + kk) * HD + c, &sK[kk][c]);
    }
    // sV: 64 d x 32 keys = 512 8-byte chunks
#pragma unroll
    for (int i = 0; i < 2; ++i) {
      const int idx = i * 256 + tid;
      const int dd = idx >> 3, c = (idx & 7) * 4;
      cp8_g2l(Vb + (size_t)dd * S_ + key0 + c, &sV[dd][c]);
    }
    if (key0 + 32 < S_) {
      __builtin_prefetch(Kb + (size_t)(key0 + 32 + (tid >> 3)) * HD + (tid & 7) * 8, 0, 0);
      __builtin_prefetch(Vb + (size_t)(tid >> 2) * S_ + key0 + 32 + (tid & 3) * 8, 0, 0);
    }
    async_fence_acquire();

    // scores: two 16-key subtiles, d-dim 64 -> 2 WMMAs each
    v8f s0 = {}, s1 = {};
    {
      v16bf b00 = load_b_bf16(&sK[ln][hf * 16]);
      v16bf b01 = load_b_bf16(&sK[ln][32 + hf * 16]);
      s0 = WMMA_BF16(aq0, b00, s0);
      s0 = WMMA_BF16(aq1, b01, s0);
      v16bf b10 = load_b_bf16(&sK[16 + ln][hf * 16]);
      v16bf b11 = load_b_bf16(&sK[16 + ln][32 + hf * 16]);
      s1 = WMMA_BF16(aq0, b10, s1);
      s1 = WMMA_BF16(aq1, b11, s1);
    }

    // online softmax (row M = v + 8*hf lives across a 16-lane half)
#pragma unroll
    for (int v = 0; v < 8; ++v) {
      float cm = fmaxf(s0[v], s1[v]);
      cm = fmaxf(cm, __shfl_xor(cm, 1, 32));
      cm = fmaxf(cm, __shfl_xor(cm, 2, 32));
      cm = fmaxf(cm, __shfl_xor(cm, 4, 32));
      cm = fmaxf(cm, __shfl_xor(cm, 8, 32));
      const float mnew = fmaxf(mrow[v], cm);
      const float cf = __expf(mrow[v] - mnew);
      mrow[v] = mnew;
      const float p0 = __expf(s0[v] - mnew);
      const float p1 = __expf(s1[v] - mnew);
      float rs = p0 + p1;
      rs += __shfl_xor(rs, 1, 32);
      rs += __shfl_xor(rs, 2, 32);
      rs += __shfl_xor(rs, 4, 32);
      rs += __shfl_xor(rs, 8, 32);
      lrow[v] = lrow[v] * cf + rs;
#pragma unroll
      for (int t = 0; t < 4; ++t) accO[t][v] *= cf;
      const int rr = v + 8 * hf;
      sP[wv][rr][ln]      = (bf16)p0;
      sP[wv][rr][16 + ln] = (bf16)p1;
    }

    // P(16x32) @ V(32x64): A from the per-wave LDS bounce, 4 WMMAs
    const v16bf ap = load_a_bf16(&sP[wv][ln][0], 0, hf);
#pragma unroll
    for (int t = 0; t < 4; ++t) {
      v16bf bvf = load_b_bf16(&sV[t * 16 + ln][hf * 16]);
      accO[t] = WMMA_BF16(ap, bvf, accO[t]);
    }
  }

  // epilogue: normalize, write attn output as bf16 rows of (B*S, E)
  const int batch = bh >> 4, h = bh & 15;
#pragma unroll
  for (int v = 0; v < 8; ++v) {
    const float inv = 1.0f / lrow[v];
    const int q = q0 + v + 8 * hf;
    bf16* orow = AO + (size_t)(batch * S_ + q) * E_ + h * HD;
#pragma unroll
    for (int t = 0; t < 4; ++t) orow[t * 16 + ln] = (bf16)(accO[t][v] * inv);
  }
}

// ---------------------------------------------------------------------------
extern "C" void kernel_launch(void* const* d_in, const int* in_sizes, int n_in,
                              void* d_out, int out_size, void* d_ws, size_t ws_size,
                              hipStream_t stream) {
  const float* query = (const float*)d_in[0];
  const float* key   = (const float*)d_in[1];
  const float* value = (const float*)d_in[2];
  const float* Wq = (const float*)d_in[3];
  const float* bq = (const float*)d_in[4];
  const float* Wk = (const float*)d_in[5];
  const float* bk = (const float*)d_in[6];
  const float* Wv = (const float*)d_in[7];
  const float* bv = (const float*)d_in[8];
  const float* Wo = (const float*)d_in[9];
  const float* bo = (const float*)d_in[10];
  const float* rel_table = (const float*)d_in[11];

  // workspace layout
  float* rel_sum = (float*)d_ws;                              // 1024*64 f32 = 256 KB
  bf16* Qbf  = (bf16*)((char*)d_ws + (1 << 18));
  bf16* Kbf  = Qbf  + (size_t)ROWS * E_;                      // 16 MB each
  bf16* Vtbf = Kbf  + (size_t)ROWS * E_;
  bf16* Abf  = Vtbf + (size_t)ROWS * E_;

  rel_sum_kernel<<<S_, HD, 0, stream>>>(rel_table, rel_sum);

  const dim3 gGemm(ROWS / 128, E_ / 64);                      // 64 x 16 blocks
  gemm_kernel<0><<<gGemm, 256, 0, stream>>>(query, nullptr, Wq, bq, nullptr, Qbf, nullptr);
  gemm_kernel<1><<<gGemm, 256, 0, stream>>>(key,   nullptr, Wk, bk, rel_sum, Kbf, nullptr);
  gemm_kernel<2><<<gGemm, 256, 0, stream>>>(value, nullptr, Wv, bv, nullptr, Vtbf, nullptr);

  attn_kernel<<<Bb * H_ * (S_ / 128), 256, 0, stream>>>(Qbf, Kbf, Vtbf, Abf);

  gemm_kernel<3><<<gGemm, 256, 0, stream>>>(nullptr, Abf, Wo, bo, nullptr, nullptr, (float*)d_out);
}
